// CapsNet_21766894256558
// MI455X (gfx1250) — matrix operand order repaired and checked by reading, not compiled, not run on hardware
//
#include <hip/hip_runtime.h>

typedef __attribute__((ext_vector_type(16))) _Float16 v16h;
typedef __attribute__((ext_vector_type(8)))  _Float16 v8h;
typedef __attribute__((ext_vector_type(8)))  float    v8f;

#define NB        256      // batch
#define M1        102400   // NB*20*20
#define K1        96       // 81 padded to 96
#define N1        256
#define M2        9216     // NB*36
#define K2        20736    // 81*256
#define N2        256
#define NROUTES   1152
#define NCAPS     10
#define LDA_S     40
#define LDB_S     40

// ---------------- WMMA fragment helpers (wave32, 16x16x32 f16) ----------------
// A (16x32): lane<16 -> M=lane, K halves {0..7,16..23}; lane>=16 -> M=lane-16, K {8..15,24..31}
static __device__ inline v16h frag_a(const _Float16* As, int lane) {
  const int m    = lane & 15;
  const int koff = (lane & 16) >> 1;          // 0 or 8
  const _Float16* p = As + m * LDA_S + koff;
  v16h a;
#pragma unroll
  for (int v = 0; v < 4; ++v) { a[2*v]   = p[2*v];     a[2*v+1]   = p[2*v+1]; }
#pragma unroll
  for (int v = 4; v < 8; ++v) { a[2*v]   = p[8 + 2*v]; a[2*v+1]   = p[8 + 2*v + 1]; }
  return a;
}
// B (32x16) staged in LDS as [n][k]: lane<16 -> N=lane, K {0..7,16..23}; lane>=16 -> N=lane-16, K {8..15,24..31}
static __device__ inline v16h frag_b(const _Float16* BsT, int lane) {
  const int n    = lane & 15;
  const int koff = (lane & 16) >> 1;          // 0 or 8
  const _Float16* p = BsT + n * LDB_S + koff;
  v16h b;
#pragma unroll
  for (int h = 0; h < 16; ++h) b[h] = p[h < 8 ? h : h + 8];
  return b;
}

// ---------------- Generic f16 GEMM: C[M,N] = act(A[M,K]*BT[N,K]^T + bias) ----------------
// BT is pre-transposed (N-major, [N][K]). 128 threads = 4 waves; 64x64 tile;
// each wave 32x32 (2x2 wmma). K % 32 == 0.
template <int ACT, int OUTF16>
__global__ __launch_bounds__(128) void gemm_f16_kernel(
    const _Float16* __restrict__ A, const _Float16* __restrict__ BT,
    const float* __restrict__ bias, void* __restrict__ Cp, int M, int N, int K) {
  __shared__ _Float16 As[64 * LDA_S];
  __shared__ _Float16 Bs[64 * LDB_S];  // [n][k]
  const int tid  = threadIdx.x;
  const int wid  = tid >> 5, lane = tid & 31;
  const int m0   = blockIdx.x * 64, n0 = blockIdx.y * 64;
  const int wr   = (wid >> 1) * 32, wc = (wid & 1) * 32;
  const int row2 = tid >> 1, akh = (tid & 1) * 16;   // staging: 64 rows x 2 half-chunks
  const int gmA  = m0 + row2;
  const int gnB  = n0 + row2;
  v8f cc[2][2] = {};
  const int KT = K >> 5;
  for (int kt = 0; kt < KT; ++kt) {
    const int k0 = kt << 5;
    { // stage A (row-major 64x32)
      v8h a0 = {}, a1 = {};
      if (gmA < M) {
        const v8h* src = (const v8h*)(A + (size_t)gmA * K + k0 + akh);
        a0 = src[0]; a1 = src[1];
      }
      v8h* dst = (v8h*)(As + row2 * LDA_S + akh);
      dst[0] = a0; dst[1] = a1;
    }
    { // stage B (N-major 64x32) -- same access pattern as A, no transpose needed
      v8h b0 = {}, b1 = {};
      if (gnB < N) {
        const v8h* src = (const v8h*)(BT + (size_t)gnB * K + k0 + akh);
        b0 = src[0]; b1 = src[1];
      }
      v8h* dst = (v8h*)(Bs + row2 * LDB_S + akh);
      dst[0] = b0; dst[1] = b1;
    }
    if (kt + 1 < KT) {  // gfx1250 global_prefetch_b8 for next K tile
      if (gmA < M) __builtin_prefetch(A + (size_t)gmA * K + k0 + 32 + akh, 0, 1);
      if (gnB < N) __builtin_prefetch(BT + (size_t)gnB * K + k0 + 32 + akh, 0, 1);
    }
    __syncthreads();
    v16h a0 = frag_a(As + wr * LDA_S, lane);
    v16h a1 = frag_a(As + (wr + 16) * LDA_S, lane);
    v16h b0 = frag_b(Bs + wc * LDB_S, lane);
    v16h b1 = frag_b(Bs + (wc + 16) * LDB_S, lane);
    cc[0][0] = __builtin_amdgcn_wmma_f32_16x16x32_f16(false, a0, false, b0, (short)0, cc[0][0], false, false);
    cc[0][1] = __builtin_amdgcn_wmma_f32_16x16x32_f16(false, a0, false, b1, (short)0, cc[0][1], false, false);
    cc[1][0] = __builtin_amdgcn_wmma_f32_16x16x32_f16(false, a1, false, b0, (short)0, cc[1][0], false, false);
    cc[1][1] = __builtin_amdgcn_wmma_f32_16x16x32_f16(false, a1, false, b1, (short)0, cc[1][1], false, false);
    __syncthreads();
  }
  // C layout: VGPR r -> M = r (+8 for lanes 16..31), N = lane&15
  const int mo = (lane & 16) >> 1, no = lane & 15;
#pragma unroll
  for (int sm = 0; sm < 2; ++sm)
#pragma unroll
    for (int sn = 0; sn < 2; ++sn) {
      int col = n0 + wc + sn * 16 + no;
      if (col >= N) continue;
      float bv = bias ? bias[col] : 0.f;
#pragma unroll
      for (int r = 0; r < 8; ++r) {
        int row = m0 + wr + sm * 16 + mo + r;
        if (row >= M) continue;
        float v = cc[sm][sn][r] + bv;
        if (ACT == 1) v = v > 0.f ? v : 0.f;
        else if (ACT == 2) v = 1.f / (1.f + __expf(-v));
        if (OUTF16) ((_Float16*)Cp)[(size_t)row * N + col] = (_Float16)v;
        else        ((float*)Cp)[(size_t)row * N + col] = v;
      }
    }
}

// ---------------- PrimaryCaps implicit-GEMM conv (256->256, 9x9, s2) ----------------
// X: f16 [NB,20,20,256] channel-last.  BwT: f16 [N2=256][K2=20736] N-major (k = tap*256+cin)
// out: f32 [NB,256,36]  (matches reshape to [B,1152,8])
__global__ __launch_bounds__(128) void pc_conv_kernel(
    const _Float16* __restrict__ X, const _Float16* __restrict__ BwT,
    const float* __restrict__ bias, float* __restrict__ out) {
  __shared__ _Float16 As[64 * LDA_S];
  __shared__ _Float16 Bs[64 * LDB_S];
  const int tid  = threadIdx.x;
  const int wid  = tid >> 5, lane = tid & 31;
  const int m0   = blockIdx.x * 64, n0 = blockIdx.y * 64;
  const int wr   = (wid >> 1) * 32, wc = (wid & 1) * 32;
  const int row2 = tid >> 1, akh = (tid & 1) * 16;
  const int gm   = m0 + row2;
  const int ab   = gm / 36, apix = gm - ab * 36;
  const int aoy  = apix / 6, aox = apix - aoy * 6;
  const _Float16* browp = BwT + (size_t)(n0 + row2) * K2;
  v8f cc[2][2] = {};
  for (int kt = 0; kt < (K2 >> 5); ++kt) {
    const int k0  = kt << 5;
    const int tap = k0 >> 8, c0 = k0 & 255;
    const int ky  = tap / 9, kx = tap - ky * 9;
    { // stage A: 32-channel chunk of input pixel (2*oy+ky, 2*ox+kx)
      const int iy = 2 * aoy + ky, ix = 2 * aox + kx;
      const v8h* src = (const v8h*)(X + (((size_t)ab * 20 + iy) * 20 + ix) * 256 + c0 + akh);
      v8h* dst = (v8h*)(As + row2 * LDA_S + akh);
      dst[0] = src[0]; dst[1] = src[1];
    }
    { // stage B (N-major)
      const v8h* src = (const v8h*)(browp + k0 + akh);
      v8h* dst = (v8h*)(Bs + row2 * LDB_S + akh);
      dst[0] = src[0]; dst[1] = src[1];
    }
    if (kt + 1 < (K2 >> 5))
      __builtin_prefetch(browp + k0 + 32 + akh, 0, 1);
    __syncthreads();
    v16h a0 = frag_a(As + wr * LDA_S, lane);
    v16h a1 = frag_a(As + (wr + 16) * LDA_S, lane);
    v16h b0 = frag_b(Bs + wc * LDB_S, lane);
    v16h b1 = frag_b(Bs + (wc + 16) * LDB_S, lane);
    cc[0][0] = __builtin_amdgcn_wmma_f32_16x16x32_f16(false, a0, false, b0, (short)0, cc[0][0], false, false);
    cc[0][1] = __builtin_amdgcn_wmma_f32_16x16x32_f16(false, a0, false, b1, (short)0, cc[0][1], false, false);
    cc[1][0] = __builtin_amdgcn_wmma_f32_16x16x32_f16(false, a1, false, b0, (short)0, cc[1][0], false, false);
    cc[1][1] = __builtin_amdgcn_wmma_f32_16x16x32_f16(false, a1, false, b1, (short)0, cc[1][1], false, false);
    __syncthreads();
  }
  const int mo = (lane & 16) >> 1, no = lane & 15;
#pragma unroll
  for (int sm = 0; sm < 2; ++sm)
#pragma unroll
    for (int sn = 0; sn < 2; ++sn) {
      int col = n0 + wc + sn * 16 + no;
      float bv = bias[col];
#pragma unroll
      for (int r = 0; r < 8; ++r) {
        int row = m0 + wr + sm * 16 + mo + r;
        int b = row / 36, pix = row - b * 36;
        out[((size_t)b * 256 + col) * 36 + pix] = cc[sm][sn][r] + bv;
      }
    }
}

// ---------------- prep / small kernels ----------------
__global__ void im2col1_kernel(const float* __restrict__ data, _Float16* __restrict__ A1) {
  int idx = blockIdx.x * 256 + threadIdx.x;
  if (idx >= M1 * K1) return;
  int m = idx / K1, k = idx - m * K1;
  int b = m / 400, pix = m - b * 400;
  int oy = pix / 20, ox = pix - oy * 20;
  float v = 0.f;
  if (k < 81) {
    int ky = k / 9, kx = k - ky * 9;
    v = data[(size_t)b * 784 + (oy + ky) * 28 + (ox + kx)];
  }
  A1[idx] = (_Float16)v;
}

// conv1_w [256,1,9,9] -> w1t_T [N1=256][K1=96]  (zero-padded taps 81..95)
__global__ void prep_w1_kernel(const float* __restrict__ w, _Float16* __restrict__ wt) {
  int idx = blockIdx.x * 256 + threadIdx.x;
  if (idx >= N1 * K1) return;
  int n = idx / K1, k = idx - n * K1;
  wt[idx] = (k < 81) ? (_Float16)w[n * 81 + k] : (_Float16)0.f;
}

// pc_w [256(out),256(in),9,9] -> pcwt_T [N2=256][K2=20736], k = tap*256 + cin
__global__ void prep_pcw_kernel(const float* __restrict__ w, _Float16* __restrict__ wt) {
  int idx = blockIdx.x * 256 + threadIdx.x;
  if (idx >= N2 * K2) return;
  int n = idx / K2, k = idx - n * K2;
  int tap = k >> 8, cin = k & 255;
  wt[idx] = (_Float16)w[((size_t)n * 256 + cin) * 81 + tap];
}

// dec weight [K,N] f32 -> [N][K] f16 (transpose + convert)
__global__ void cvt_transpose_f16_kernel(const float* __restrict__ s, _Float16* __restrict__ d,
                                         int K, int N) {
  int idx = blockIdx.x * 256 + threadIdx.x;
  if (idx >= K * N) return;
  int n = idx / K, k = idx - n * K;
  d[idx] = (_Float16)s[(size_t)k * N + n];
}

__global__ void fill_zero_kernel(float* p, int n) {
  int i = blockIdx.x * 256 + threadIdx.x;
  if (i < n) p[i] = 0.f;
}

__global__ void squash_u_kernel(const float* __restrict__ pcout, float* __restrict__ u) {
  int idx = blockIdx.x * 256 + threadIdx.x;   // b*1152 + r
  const float* p = pcout + (size_t)idx * 8;
  float x[8], sq = 0.f;
#pragma unroll
  for (int i = 0; i < 8; ++i) { x[i] = p[i]; sq += x[i] * x[i]; }
  float scale = sq / ((1.f + sq) * sqrtf(sq) + 1e-30f);
#pragma unroll
  for (int i = 0; i < 8; ++i) u[(size_t)idx * 8 + i] = x[i] * scale;
}

// u_hat[b,r,c,o] = sum_i W[r,c,o,i] * u[b,r,i]; block = route r, 160 threads (t = c*16+o)
__global__ __launch_bounds__(160) void uhat_kernel(
    const float* __restrict__ W, const float* __restrict__ u, float* __restrict__ uhat) {
  __shared__ float Wl[1280];
  __shared__ float ul[NB * 8];
  const int r = blockIdx.x, t = threadIdx.x;
  for (int i = t; i < 1280; i += 160) Wl[i] = W[(size_t)r * 1280 + i];
  for (int i = t; i < NB * 8; i += 160) {
    int b = i >> 3, ii = i & 7;
    ul[i] = u[((size_t)b * NROUTES + r) * 8 + ii];
  }
  __syncthreads();
  for (int b = 0; b < NB; ++b) {
    float acc = 0.f;
#pragma unroll
    for (int i = 0; i < 8; ++i) acc += ul[b * 8 + i] * Wl[t * 8 + i];
    uhat[((size_t)b * NROUTES + r) * 160 + t] = acc;
  }
}

__global__ void softmax_routes_kernel(const float* __restrict__ bij, float* __restrict__ cij) {
  __shared__ float red[256];
  const int c = blockIdx.x, t = threadIdx.x;
  float m = -1e30f;
  for (int r = t; r < NROUTES; r += 256) m = fmaxf(m, bij[r * NCAPS + c]);
  red[t] = m; __syncthreads();
  for (int s = 128; s > 0; s >>= 1) { if (t < s) red[t] = fmaxf(red[t], red[t + s]); __syncthreads(); }
  float mx = red[0]; __syncthreads();
  float sum = 0.f;
  for (int r = t; r < NROUTES; r += 256) sum += __expf(bij[r * NCAPS + c] - mx);
  red[t] = sum; __syncthreads();
  for (int s = 128; s > 0; s >>= 1) { if (t < s) red[t] += red[t + s]; __syncthreads(); }
  float inv = 1.f / red[0];
  for (int r = t; r < NROUTES; r += 256) cij[r * NCAPS + c] = __expf(bij[r * NCAPS + c] - mx) * inv;
}

// s_j = sum_r c[r,c]*u_hat[b,r,c,:]; v = squash(s_j). block = b*10+c, 256 threads
__global__ void s_v_kernel(const float* __restrict__ cij, const float* __restrict__ uhat,
                           float* __restrict__ vbuf) {
  __shared__ float red[256];
  __shared__ float scale_s;
  const int bc = blockIdx.x, b = bc / NCAPS, c = bc - b * NCAPS;
  const int t = threadIdx.x, o = t & 15, rs = t >> 4;
  float acc = 0.f;
  for (int r = rs; r < NROUTES; r += 16)
    acc += cij[r * NCAPS + c] * uhat[((size_t)b * NROUTES + r) * 160 + c * 16 + o];
  red[t] = acc; __syncthreads();
  for (int s = 128; s >= 16; s >>= 1) { if (t < s) red[t] += red[t + s]; __syncthreads(); }
  if (t == 0) {
    float sq = 0.f;
    for (int o2 = 0; o2 < 16; ++o2) sq += red[o2] * red[o2];
    scale_s = sq / ((1.f + sq) * sqrtf(sq) + 1e-30f);
  }
  __syncthreads();
  if (t < 16) vbuf[(size_t)bc * 16 + t] = red[t] * scale_s;
}

// b_ij[r,c] += (1/B) * sum_{b,o} u_hat[b,r,c,o]*v[b,c,o]. block = r*10+c
__global__ void update_b_kernel(const float* __restrict__ uhat, const float* __restrict__ vbuf,
                                float* __restrict__ bij) {
  __shared__ float red[256];
  const int rc = blockIdx.x, r = rc / NCAPS, c = rc - r * NCAPS;
  const int t = threadIdx.x, o = t & 15, bs = t >> 4;
  float acc = 0.f;
  for (int b = bs; b < NB; b += 16)
    acc += uhat[((size_t)b * NROUTES + r) * 160 + c * 16 + o] * vbuf[(b * NCAPS + c) * 16 + o];
  red[t] = acc; __syncthreads();
  for (int s = 128; s > 0; s >>= 1) { if (t < s) red[t] += red[t + s]; __syncthreads(); }
  if (t == 0) bij[rc] += red[0] * (1.f / (float)NB);
}

__global__ void norms_kernel(const float* __restrict__ vbuf, float* __restrict__ cls) {
  int i = blockIdx.x * 256 + threadIdx.x;   // b*10+c
  if (i >= NB * NCAPS) return;
  float s = 0.f;
#pragma unroll
  for (int o = 0; o < 16; ++o) { float x = vbuf[(size_t)i * 16 + o]; s += x * x; }
  cls[i] = sqrtf(s);
}

// softmax over batch axis (as in reference), per class column
__global__ void softmax_batch_kernel(const float* __restrict__ cls, float* __restrict__ csm) {
  __shared__ float red[256];
  const int c = blockIdx.x, t = threadIdx.x;
  float x = cls[t * NCAPS + c];
  red[t] = x; __syncthreads();
  for (int s = 128; s > 0; s >>= 1) { if (t < s) red[t] = fmaxf(red[t], red[t + s]); __syncthreads(); }
  float mx = red[0]; __syncthreads();
  red[t] = __expf(x - mx); __syncthreads();
  for (int s = 128; s > 0; s >>= 1) { if (t < s) red[t] += red[t + s]; __syncthreads(); }
  csm[t * NCAPS + c] = __expf(x - mx) / red[0];
}

__global__ __launch_bounds__(192) void mask_out_kernel(
    const float* __restrict__ csm, const float* __restrict__ vbuf,
    float* __restrict__ out_v, float* __restrict__ out_mask, _Float16* __restrict__ tf16) {
  __shared__ int ci_s;
  const int b = blockIdx.x, t = threadIdx.x;
  if (t == 0) {
    int best = 0; float bv = csm[b * NCAPS];
    for (int c = 1; c < NCAPS; ++c) { float v = csm[b * NCAPS + c]; if (v > bv) { bv = v; best = c; } }
    ci_s = best;
  }
  __syncthreads();
  const int ci = ci_s;
  if (t < 160) {
    float v = vbuf[(size_t)b * 160 + t];
    out_v[(size_t)b * 160 + t] = v;
    tf16[(size_t)b * 160 + t] = (_Float16)(((t >> 4) == ci) ? v : 0.f);
  }
  if (t < NCAPS) out_mask[b * NCAPS + t] = (t == ci) ? 1.f : 0.f;
}

// ---------------- host ----------------
extern "C" void kernel_launch(void* const* d_in, const int* in_sizes, int n_in,
                              void* d_out, int out_size, void* d_ws, size_t ws_size,
                              hipStream_t stream) {
  const float* data    = (const float*)d_in[0];
  const float* conv1_w = (const float*)d_in[1];
  const float* conv1_b = (const float*)d_in[2];
  const float* pc_w    = (const float*)d_in[3];
  const float* pc_b    = (const float*)d_in[4];
  const float* W_dc    = (const float*)d_in[5];
  const float* dec_w1  = (const float*)d_in[6];
  const float* dec_b1  = (const float*)d_in[7];
  const float* dec_w2  = (const float*)d_in[8];
  const float* dec_b2  = (const float*)d_in[9];
  const float* dec_w3  = (const float*)d_in[10];
  const float* dec_b3  = (const float*)d_in[11];

  char* ws = (char*)d_ws;
  size_t off = 0;
  auto alloc = [&](size_t bytes) { size_t r = off; off += (bytes + 255) & ~(size_t)255; return r; };
  _Float16* a1    = (_Float16*)(ws + alloc((size_t)M1 * K1 * 2));
  _Float16* w1t   = (_Float16*)(ws + alloc((size_t)N1 * K1 * 2));     // [N][K]
  _Float16* x16   = (_Float16*)(ws + alloc((size_t)M1 * N1 * 2));
  _Float16* pcwt  = (_Float16*)(ws + alloc((size_t)N2 * K2 * 2));     // [N][K]
  float*    pcout = (float*)   (ws + alloc((size_t)NB * 9216 * 4));
  float*    u     = (float*)   (ws + alloc((size_t)NB * 9216 * 4));
  float*    uhat  = (float*)   (ws + alloc((size_t)NB * NROUTES * 160 * 4));
  float*    bij   = (float*)   (ws + alloc((size_t)NROUTES * NCAPS * 4));
  float*    cij   = (float*)   (ws + alloc((size_t)NROUTES * NCAPS * 4));
  float*    vbuf  = (float*)   (ws + alloc((size_t)NB * 160 * 4));
  float*    cls   = (float*)   (ws + alloc((size_t)NB * NCAPS * 4));
  float*    csm   = (float*)   (ws + alloc((size_t)NB * NCAPS * 4));
  _Float16* tf16  = (_Float16*)(ws + alloc((size_t)NB * 160 * 2));
  _Float16* h1    = (_Float16*)(ws + alloc((size_t)NB * 512 * 2));
  _Float16* h2    = (_Float16*)(ws + alloc((size_t)NB * 1024 * 2));
  _Float16* dw1t  = (_Float16*)(ws + alloc((size_t)512 * 160 * 2));   // [N][K]
  _Float16* dw2t  = (_Float16*)(ws + alloc((size_t)1024 * 512 * 2));  // [N][K]
  _Float16* dw3t  = (_Float16*)(ws + alloc((size_t)784 * 1024 * 2));  // [N][K]

  float* outp     = (float*)d_out;
  float* out_v    = outp;                 // [256,10,16,1]
  float* recon    = outp + 40960;         // [256,1,28,28]
  float* out_mask = outp + 241664;        // [256,10]

  // Conv1 as GEMM
  im2col1_kernel<<<(M1 * K1 + 255) / 256, 256, 0, stream>>>(data, a1);
  prep_w1_kernel<<<(N1 * K1 + 255) / 256, 256, 0, stream>>>(conv1_w, w1t);
  gemm_f16_kernel<1, 1><<<dim3(M1 / 64, N1 / 64), 128, 0, stream>>>(
      a1, w1t, conv1_b, (void*)x16, M1, N1, K1);

  // PrimaryCaps conv as implicit GEMM
  prep_pcw_kernel<<<((N2 * K2) + 255) / 256, 256, 0, stream>>>(pc_w, pcwt);
  pc_conv_kernel<<<dim3(M2 / 64, N2 / 64), 128, 0, stream>>>(x16, pcwt, pc_b, pcout);

  // squash -> u ; u_hat
  squash_u_kernel<<<NROUTES, 256, 0, stream>>>(pcout, u);
  uhat_kernel<<<NROUTES, 160, 0, stream>>>(W_dc, u, uhat);

  // dynamic routing (3 iterations)
  fill_zero_kernel<<<(NROUTES * NCAPS + 255) / 256, 256, 0, stream>>>(bij, NROUTES * NCAPS);
  for (int it = 0; it < 3; ++it) {
    softmax_routes_kernel<<<NCAPS, 256, 0, stream>>>(bij, cij);
    s_v_kernel<<<NB * NCAPS, 256, 0, stream>>>(cij, uhat, vbuf);
    if (it < 2)
      update_b_kernel<<<NROUTES * NCAPS, 256, 0, stream>>>(uhat, vbuf, bij);
  }

  // classes / mask / masked input for decoder; emit output + masked
  norms_kernel<<<(NB * NCAPS + 255) / 256, 256, 0, stream>>>(vbuf, cls);
  softmax_batch_kernel<<<NCAPS, 256, 0, stream>>>(cls, csm);
  mask_out_kernel<<<NB, 192, 0, stream>>>(csm, vbuf, out_v, out_mask, tf16);

  // decoder (weights pre-transposed to [N][K] f16)
  cvt_transpose_f16_kernel<<<(160 * 512 + 255) / 256, 256, 0, stream>>>(dec_w1, dw1t, 160, 512);
  cvt_transpose_f16_kernel<<<(512 * 1024 + 255) / 256, 256, 0, stream>>>(dec_w2, dw2t, 512, 1024);
  cvt_transpose_f16_kernel<<<(1024 * 784 + 255) / 256, 256, 0, stream>>>(dec_w3, dw3t, 1024, 784);
  gemm_f16_kernel<1, 1><<<dim3(NB / 64, 512 / 64), 128, 0, stream>>>(
      tf16, dw1t, dec_b1, (void*)h1, NB, 512, 160);
  gemm_f16_kernel<1, 1><<<dim3(NB / 64, 1024 / 64), 128, 0, stream>>>(
      h1, dw2t, dec_b2, (void*)h2, NB, 1024, 512);
  gemm_f16_kernel<2, 0><<<dim3(NB / 64, (784 + 63) / 64), 128, 0, stream>>>(
      h2, dw3t, dec_b3, (void*)recon, NB, 784, 1024);
}
